// GNN_6382321402118
// MI455X (gfx1250) — matrix operand order repaired and checked by reading, not compiled, not run on hardware
//
#include <hip/hip_runtime.h>

// ---------------- problem constants (match reference) ----------------
static constexpr int NODES = 50000;
static constexpr int INC   = 256;   // in_channels
static constexpr int HEADS = 4;
static constexpr int OUTD  = 64;    // per-head dim
static constexpr int HD    = 256;   // HEADS*OUTD
static constexpr int RREL  = 5;
static constexpr int EDGES = 800000;
static constexpr int NRH   = NODES * RREL * HEADS; // 1,000,000 segments*heads

// ---------------- WMMA types ----------------
typedef __attribute__((ext_vector_type(16))) __bf16 v16bf;
typedef __attribute__((ext_vector_type(8)))  float  v8f;
typedef __attribute__((ext_vector_type(4)))  unsigned int u32x4;

union FragBF { u32x4 u[2]; v16bf v; };

__device__ inline unsigned short f2bf(float f) {
    unsigned int u = __float_as_uint(f);
    unsigned int r = u + 0x7FFFu + ((u >> 16) & 1u);   // round-to-nearest-even
    return (unsigned short)(r >> 16);
}

// ---------------- kernel 1: x fp32 -> bf16 ----------------
__global__ __launch_bounds__(256) void cvt_x_kernel(const float* __restrict__ x,
                                                    unsigned short* __restrict__ xb,
                                                    int n) {
    int i = blockIdx.x * 256 + threadIdx.x;
    if (i < n) xb[i] = f2bf(x[i]);
}

// ---------------- kernel 2: W [R][IN][HD] fp32 -> Wt bf16 [3][R][HD][IN] ----------------
__global__ __launch_bounds__(256) void cvt_w_kernel(const float* __restrict__ Wq,
                                                    const float* __restrict__ Wk,
                                                    const float* __restrict__ Wv,
                                                    unsigned short* __restrict__ Wb) {
    int which = blockIdx.y;
    const float* W = (which == 0) ? Wq : (which == 1) ? Wk : Wv;
    int i = blockIdx.x * 256 + threadIdx.x;          // over R*IN*HD
    if (i >= RREL * INC * HD) return;
    int r   = i / (INC * HD);
    int rem = i - r * (INC * HD);
    int c   = rem / HD;
    int h   = rem - c * HD;
    Wb[(((size_t)which * RREL + r) * HD + h) * INC + c] = f2bf(W[i]);
}

// ---------------- kernel 3: init softmax max (-inf), sums (0), out (bias) ----------------
__global__ __launch_bounds__(256) void init_kernel(unsigned int* __restrict__ mbits,
                                                   float* __restrict__ sbuf,
                                                   float* __restrict__ out,
                                                   const float* __restrict__ bias) {
    int i = blockIdx.x * 256 + threadIdx.x;
    if (i < NRH) { mbits[i] = 0xFF800000u; sbuf[i] = 0.0f; }
    if (i < NODES * HD) out[i] = bias[i & (HD - 1)];
}

// ---------------- kernel 4: bf16 WMMA GEMM, QKV[which][r] = xb @ Wt + b ----------------
// one wave computes a 16(M) x 64(N) tile; K = 256 in steps of 32
__global__ __launch_bounds__(256) void gemm_qkv_kernel(const unsigned short* __restrict__ xb,   // [N][IN] bf16
                                                       const unsigned short* __restrict__ Wb,   // [3][R][HD][IN] bf16
                                                       const float* __restrict__ bq,
                                                       const float* __restrict__ bk,
                                                       const float* __restrict__ bv,
                                                       float* __restrict__ QKV) {               // [3][R][N][HD]
    const int tiles_m = NODES / 16;        // 3125
    const int tiles_n = HD / 64;           // 4
    const int per_rm  = tiles_m * tiles_n; // 12500
    const int total   = 3 * RREL * per_rm; // 187500 waves

    int wid = blockIdx.x * 8 + (threadIdx.x >> 5);
    if (wid >= total) return;

    int which = wid / (RREL * per_rm);
    int rem   = wid - which * (RREL * per_rm);
    int r     = rem / per_rm;
    int t     = rem - r * per_rm;
    int tm    = t / tiles_n;
    int tn    = t - tm * tiles_n;

    int lane = threadIdx.x & 31;
    int hk   = lane >> 4;      // half-wave selector
    int l15  = lane & 15;

    // A fragment source: row (tm*16 + l15); lanes<16 start at k+0, lanes>=16 at k+8
    const unsigned short* arow  = xb + (size_t)(tm * 16 + l15) * INC + 8 * hk;
    const unsigned short* wbase = Wb + ((size_t)(which * RREL + r) * HD) * INC;
    int col0 = tn * 64;

    v8f acc[4] = {};
    for (int k = 0; k < INC; k += 32) {
        FragBF fa;
        const u32x4* pa = (const u32x4*)(arow + k);
        fa.u[0] = pa[0];   // K k..k+7   (this half)
        fa.u[1] = pa[2];   // K k+16..k+23 (this half)
        #pragma unroll
        for (int j = 0; j < 4; ++j) {
            FragBF fb;
            const u32x4* pb = (const u32x4*)(wbase + (size_t)(col0 + j * 16 + l15) * INC + k + 8 * hk);
            fb.u[0] = pb[0];
            fb.u[1] = pb[2];
            acc[j] = __builtin_amdgcn_wmma_f32_16x16x32_bf16(
                false, fa.v, false, fb.v, (short)0, acc[j], false, false);
        }
    }

    const float* bias = (which == 0) ? bq : (which == 1) ? bk : bv;
    float* outb = QKV + ((size_t)(which * RREL + r) * NODES) * HD;
    #pragma unroll
    for (int j = 0; j < 4; ++j) {
        int col = col0 + j * 16 + l15;
        float bb = bias[r * HD + col];
        #pragma unroll
        for (int i = 0; i < 8; ++i) {
            int row = tm * 16 + i + 8 * hk;       // C/D layout: M = i (+8 for upper half)
            outb[(size_t)row * HD + col] = acc[j][i] + bb;
        }
    }
}

// ---------------- kernel 5: per-edge scores + segment max ----------------
// one wave per edge; lane owns 8 contiguous floats; head = lane>>3 (64 floats/head)
__global__ __launch_bounds__(256) void score_kernel(const float* __restrict__ QKV,
                                                    const int* __restrict__ ei,
                                                    const int* __restrict__ et,
                                                    float* __restrict__ score,
                                                    float* __restrict__ mbuf) {
    int e = blockIdx.x * 8 + (threadIdx.x >> 5);
    if (e >= EDGES) return;
    int lane = threadIdx.x & 31;
    int src = ei[e];
    int dst = ei[EDGES + e];
    int r   = et[e];

    const float* Q = QKV + ((size_t)(0 * RREL + r) * NODES + dst) * HD + lane * 8;
    const float* K = QKV + ((size_t)(1 * RREL + r) * NODES + src) * HD + lane * 8;
    float4 q0 = *(const float4*)Q;
    float4 q1 = *(const float4*)(Q + 4);
    float4 k0 = *(const float4*)K;
    float4 k1 = *(const float4*)(K + 4);
    float d = q0.x * k0.x + q0.y * k0.y + q0.z * k0.z + q0.w * k0.w
            + q1.x * k1.x + q1.y * k1.y + q1.z * k1.z + q1.w * k1.w;
    // reduce across the 8 lanes of this head
    d += __shfl_xor(d, 1);
    d += __shfl_xor(d, 2);
    d += __shfl_xor(d, 4);
    if ((lane & 7) == 0) {
        int h = lane >> 3;
        float sc = d * 0.125f;                  // OUT^-0.5 = 1/8
        score[(size_t)e * HEADS + h] = sc;
        int seg = dst * RREL + r;
        float* addr = mbuf + (size_t)seg * HEADS + h;
        // order-independent float max via sign-split integer atomics
        if (sc >= 0.0f) atomicMax((int*)addr, __float_as_int(sc));
        else            atomicMin((unsigned int*)addr, __float_as_uint(sc));
    }
}

// ---------------- kernel 6: exp(score - max), segment sum ----------------
__global__ __launch_bounds__(256) void expsum_kernel(const int* __restrict__ ei,
                                                     const int* __restrict__ et,
                                                     const float* __restrict__ mbuf,
                                                     float* __restrict__ score,
                                                     float* __restrict__ sbuf) {
    int i = blockIdx.x * 256 + threadIdx.x;
    if (i >= EDGES * HEADS) return;
    int e = i >> 2;
    int h = i & 3;
    int dst = ei[EDGES + e];
    int r   = et[e];
    int seg = dst * RREL + r;
    float v = __expf(score[i] - mbuf[(size_t)seg * HEADS + h]);
    score[i] = v;
    atomicAdd(sbuf + (size_t)seg * HEADS + h, v);
}

// ---------------- kernel 7: alpha-weighted scatter of V into out ----------------
__global__ __launch_bounds__(256) void scatter_kernel(const float* __restrict__ QKV,
                                                      const int* __restrict__ ei,
                                                      const int* __restrict__ et,
                                                      const float* __restrict__ score,
                                                      const float* __restrict__ sbuf,
                                                      float* __restrict__ out) {
    int e = blockIdx.x * 8 + (threadIdx.x >> 5);
    if (e >= EDGES) return;
    int lane = threadIdx.x & 31;
    int src = ei[e];
    int dst = ei[EDGES + e];
    int r   = et[e];
    int h   = lane >> 3;
    int seg = dst * RREL + r;
    float alpha = score[(size_t)e * HEADS + h] / sbuf[(size_t)seg * HEADS + h];

    const float* V = QKV + ((size_t)(2 * RREL + r) * NODES + src) * HD + lane * 8;
    float* o = out + (size_t)dst * HD + lane * 8;
    float4 v0 = *(const float4*)V;
    float4 v1 = *(const float4*)(V + 4);
    atomicAdd(o + 0, v0.x * alpha);
    atomicAdd(o + 1, v0.y * alpha);
    atomicAdd(o + 2, v0.z * alpha);
    atomicAdd(o + 3, v0.w * alpha);
    atomicAdd(o + 4, v1.x * alpha);
    atomicAdd(o + 5, v1.y * alpha);
    atomicAdd(o + 6, v1.z * alpha);
    atomicAdd(o + 7, v1.w * alpha);
}

// ---------------- launch ----------------
extern "C" void kernel_launch(void* const* d_in, const int* in_sizes, int n_in,
                              void* d_out, int out_size, void* d_ws, size_t ws_size,
                              hipStream_t stream) {
    const float* x    = (const float*)d_in[0];
    const int*   ei   = (const int*)d_in[1];   // [2][E]: row0=src, row1=dst
    const int*   et   = (const int*)d_in[2];
    const float* Wq   = (const float*)d_in[3];
    const float* bq   = (const float*)d_in[4];
    const float* Wk   = (const float*)d_in[5];
    const float* bk   = (const float*)d_in[6];
    const float* Wv   = (const float*)d_in[7];
    const float* bv   = (const float*)d_in[8];
    const float* bias = (const float*)d_in[9];
    float* out = (float*)d_out;

    // workspace carve-out (256B aligned slices)
    char* ws = (char*)d_ws;
    size_t off = 0;
    auto carve = [&](size_t bytes) -> void* {
        void* p = ws + off;
        off = (off + bytes + 255) & ~(size_t)255;
        return p;
    };
    unsigned short* xb   = (unsigned short*)carve((size_t)NODES * INC * sizeof(unsigned short));
    unsigned short* Wb   = (unsigned short*)carve((size_t)3 * RREL * HD * INC * sizeof(unsigned short));
    float*          QKV  = (float*)carve((size_t)3 * RREL * NODES * HD * sizeof(float));
    float*          scor = (float*)carve((size_t)EDGES * HEADS * sizeof(float));
    float*          mbuf = (float*)carve((size_t)NRH * sizeof(float));
    float*          sbuf = (float*)carve((size_t)NRH * sizeof(float));

    // 1) conversions + init
    {
        int n = NODES * INC;
        cvt_x_kernel<<<(n + 255) / 256, 256, 0, stream>>>(x, xb, n);
    }
    {
        dim3 g((RREL * INC * HD + 255) / 256, 3);
        cvt_w_kernel<<<g, 256, 0, stream>>>(Wq, Wk, Wv, Wb);
    }
    init_kernel<<<(NODES * HD + 255) / 256, 256, 0, stream>>>((unsigned int*)mbuf, sbuf, out, bias);

    // 2) QKV projections via bf16 WMMA (187500 waves, 8 waves/block)
    {
        int waves  = 3 * RREL * (NODES / 16) * (HD / 64);
        int blocks = (waves + 7) / 8;
        gemm_qkv_kernel<<<blocks, 256, 0, stream>>>(xb, Wb, bq, bk, bv, QKV);
    }

    // 3) edge scores + segment max (wave per edge)
    score_kernel<<<(EDGES + 7) / 8, 256, 0, stream>>>(QKV, ei, et, scor, mbuf);

    // 4) exp + segment sum
    expsum_kernel<<<(EDGES * HEADS + 255) / 256, 256, 0, stream>>>(ei, et, mbuf, scor, sbuf);

    // 5) weighted scatter into out (pre-initialized with bias)
    scatter_kernel<<<(EDGES + 7) / 8, 256, 0, stream>>>(QKV, ei, et, scor, sbuf, out);
}
